// SA_73495480369173
// MI455X (gfx1250) — matrix operand order repaired and checked by reading, not compiled, not run on hardware
//
#include <hip/hip_runtime.h>
#include <hip/hip_bf16.h>

typedef __attribute__((ext_vector_type(16))) _Float16 v16h;
typedef __attribute__((ext_vector_type(8)))  float    v8f;

#define BN_EPS 1e-5f
#define NPTS   8192
#define NG     512
#define MNB    32
#define ROWS   64    // 2 groups x 32 neighbors per block
#define STR1   136   // 128 + 8 pad (halves)
#define STRF   520   // 512 + 8 pad (halves)

union AFrag { v16h v; uint4 q[2]; };
union Acc8  { v8f  v; float f[8]; };

// ---- WMMA fragment loaders (CDNA5 16-bit layouts, wave32) -------------------
// A 16x32 f16: lane L -> row m0 + (L&15); halves e<8 => K = k0 + 8*(L>>4) + e,
// e>=8 => K = k0 + 8*(L>>4) + 16 + (e-8). Two 16-byte LDS loads.
__device__ __forceinline__ v16h load_a16(const _Float16* base, int stride,
                                         int m0, int k0, int lane) {
  const int lm = lane & 15, lg = lane >> 4;
  const _Float16* p = base + (m0 + lm) * stride + k0 + lg * 8;
  AFrag f;
  f.q[0] = *(const uint4*)(p);
  f.q[1] = *(const uint4*)(p + 16);
  return f.v;
}

// B 32x16 f16 from row-major W[nOut][K] (B[k][n] = W[n][k]):
// lane L -> col n0 + (L&15); same K layout as A: e<8 => K = k0 + 8*(L>>4) + e, e>=8 => +16.
__device__ __forceinline__ v16h load_b16(const _Float16* W, int K,
                                         int n0, int k0, int lane) {
  const int lm = lane & 15, kg = lane >> 4;
  const _Float16* p = W + (size_t)(n0 + lm) * K + k0 + kg * 8;
  AFrag f;
  f.q[0] = *(const uint4*)(p);
  f.q[1] = *(const uint4*)(p + 16);
  return f.v;
}

__device__ __forceinline__ v8f wmma_f16(v16h a, v16h b, v8f c) {
  return __builtin_amdgcn_wmma_f32_16x16x32_f16(false, a, false, b,
                                                (short)0, c, false, false);
}

// Register-blocked GEMM: per k-step load 4 A fragments once, then for each of
// NT owned n-tiles load one B fragment and issue 4 WMMAs (A reused NT times from
// registers, B reused 4x across the M=64 rows of the two batched groups).
template <int NT, int NKS>
__device__ __forceinline__ void gemm_block(const _Float16* Alds, int strideA,
                                           const _Float16* W, int K,
                                           int ntBase, int lane,
                                           Acc8 acc[NT][4]) {
#pragma unroll
  for (int j = 0; j < NT; ++j)
#pragma unroll
    for (int mt = 0; mt < 4; ++mt)
#pragma unroll
      for (int v = 0; v < 8; ++v) acc[j][mt].f[v] = 0.0f;

#pragma unroll 2
  for (int ks = 0; ks < NKS; ++ks) {
    v16h a[4];
#pragma unroll
    for (int mt = 0; mt < 4; ++mt)
      a[mt] = load_a16(Alds, strideA, mt * 16, ks * 32, lane);
#pragma unroll
    for (int j = 0; j < NT; ++j) {
      const v16h bf = load_b16(W, K, (ntBase + j) * 16, ks * 32, lane);
#pragma unroll
      for (int mt = 0; mt < 4; ++mt)
        acc[j][mt].v = wmma_f16(a[mt], bf, acc[j][mt].v);
    }
  }
}

// ---- Kernel 0: weight f32 -> f16 -------------------------------------------
__global__ __launch_bounds__(256) void convert_weights(
    const float* __restrict__ w2, const float* __restrict__ w3,
    const float* __restrict__ w4, _Float16* __restrict__ w2h,
    _Float16* __restrict__ w3h, _Float16* __restrict__ w4h) {
  int i = blockIdx.x * 256 + threadIdx.x;           // grid covers 512*512
  if (i < 256 * 128) w2h[i] = (_Float16)w2[i];
  if (i < 512 * 512) w3h[i] = (_Float16)w3[i];
  if (i < 384 * 512) w4h[i] = (_Float16)w4[i];
}

// ---- Kernel 1: farthest point sampling (one block per cloud) ----------------
__global__ __launch_bounds__(256) void fps_kernel(
    const float* __restrict__ xyz, int* __restrict__ fidx,
    float* __restrict__ center) {
  const int b = blockIdx.x, t = threadIdx.x;
  const float* P = xyz + (size_t)b * NPTS * 3;
  float dist[32];
#pragma unroll
  for (int i = 0; i < 32; ++i) dist[i] = 1e10f;
  __shared__ float redV[256];
  __shared__ int   redI[256];
  __shared__ float sF[3];
  int far = 0;
  for (int g = 0; g < NG; ++g) {
    if (t == 0) {
      fidx[b * NG + g] = far;
      float fx = P[far * 3 + 0], fy = P[far * 3 + 1], fz = P[far * 3 + 2];
      float* c = center + ((size_t)b * NG + g) * 3;
      c[0] = fx; c[1] = fy; c[2] = fz;
      sF[0] = fx; sF[1] = fy; sF[2] = fz;
    }
    __syncthreads();
    const float fx = sF[0], fy = sF[1], fz = sF[2];
    float bv = -1.0f; int bi = 1 << 30;
#pragma unroll
    for (int i = 0; i < 32; ++i) {
      const int idx = i * 256 + t;
      const float dx = P[idx * 3 + 0] - fx;
      const float dy = P[idx * 3 + 1] - fy;
      const float dz = P[idx * 3 + 2] - fz;
      const float d  = dx * dx + dy * dy + dz * dz;
      const float nd = fminf(dist[i], d);
      dist[i] = nd;
      if (nd > bv || (nd == bv && idx < bi)) { bv = nd; bi = idx; }
    }
    redV[t] = bv; redI[t] = bi;
    __syncthreads();
    for (int s = 128; s > 0; s >>= 1) {
      if (t < s) {
        const float v2 = redV[t + s]; const int i2 = redI[t + s];
        if (v2 > redV[t] || (v2 == redV[t] && i2 < redI[t])) {
          redV[t] = v2; redI[t] = i2;
        }
      }
      __syncthreads();
    }
    far = redI[0];
    __syncthreads();
  }
}

// ---- Kernel 2: 32-NN per group (one block per group) ------------------------
__global__ __launch_bounds__(256) void knn_kernel(
    const float* __restrict__ xyz, const float* __restrict__ center,
    int* __restrict__ knn) {
  const int blk = blockIdx.x, b = blk >> 9, t = threadIdx.x;
  const float* P = xyz + (size_t)b * NPTS * 3;
  const float* C = center + (size_t)blk * 3;
  __shared__ float sd2[NPTS];
  __shared__ float redV[256];
  __shared__ int   redI[256];
  const float cx = C[0], cy = C[1], cz = C[2];
  const float cn = cx * cx + cy * cy + cz * cz;
  for (int i = t; i < NPTS; i += 256) {
    const float x = P[i * 3 + 0], y = P[i * 3 + 1], z = P[i * 3 + 2];
    sd2[i] = cn + (x * x + y * y + z * z) - 2.0f * (cx * x + cy * y + cz * z);
  }
  __syncthreads();
  for (int j = 0; j < MNB; ++j) {
    float bv = 3.0e38f; int bi = 1 << 30;
#pragma unroll
    for (int i = 0; i < 32; ++i) {
      const int idx = i * 256 + t;
      const float v = sd2[idx];
      if (v < bv || (v == bv && idx < bi)) { bv = v; bi = idx; }
    }
    redV[t] = bv; redI[t] = bi;
    __syncthreads();
    for (int s = 128; s > 0; s >>= 1) {
      if (t < s) {
        const float v2 = redV[t + s]; const int i2 = redI[t + s];
        if (v2 < redV[t] || (v2 == redV[t] && i2 < redI[t])) {
          redV[t] = v2; redI[t] = i2;
        }
      }
      __syncthreads();
    }
    const int w = redI[0];
    if (t == 0) { knn[(size_t)blk * MNB + j] = w; sd2[w] = 3.4e38f; }
    __syncthreads();
  }
}

// ---- Kernel 3: fused MLP encoder, 2 groups per block (M=64), 8 waves --------
__global__ __launch_bounds__(256) void sa_mlp_kernel(
    const float* __restrict__ xyz, const float* __restrict__ center,
    const int* __restrict__ knn,
    const float* __restrict__ w1, const float* __restrict__ b1,
    const float* __restrict__ g1, const float* __restrict__ be1,
    const float* __restrict__ m1, const float* __restrict__ v1,
    const float* __restrict__ b2, const float* __restrict__ b3,
    const float* __restrict__ g3, const float* __restrict__ be3,
    const float* __restrict__ m3, const float* __restrict__ v3,
    const float* __restrict__ b4,
    const _Float16* __restrict__ w2h, const _Float16* __restrict__ w3h,
    const _Float16* __restrict__ w4h, float* __restrict__ tokens) {
  const int blk2 = blockIdx.x;                  // 0..4095
  const int gA   = blk2 * 2;                    // first group id (global)
  const int b    = gA >> 9;                     // batch (both groups same batch)
  const int t    = threadIdx.x;
  const int lane = t & 31, wave = t >> 5, lg = (t >> 4) & 1;

  __shared__ __align__(16) float    sNbr[ROWS][3];
  __shared__ __align__(16) _Float16 sA1[ROWS * STR1];     // 17 KB
  __shared__ __align__(16) _Float16 sFeat[ROWS * STRF];   // 65 KB
  __shared__ __align__(16) _Float16 sA3[ROWS * STRF];     // 65 KB

  // gather + recenter two neighborhoods (rows 0..31 = gA, 32..63 = gA+1)
  const float* P = xyz + (size_t)b * NPTS * 3;
  if (t < ROWS) {
    const int grp = gA + (t >> 5);
    const float* C = center + (size_t)grp * 3;
    const int idx = knn[(size_t)grp * MNB + (t & 31)];
    sNbr[t][0] = P[idx * 3 + 0] - C[0];
    sNbr[t][1] = P[idx * 3 + 1] - C[1];
    sNbr[t][2] = P[idx * 3 + 2] - C[2];
  }
  __syncthreads();

  // layer1: 3 -> 128, BN + ReLU (scalar; K too small for WMMA)
  for (int j = t; j < ROWS * 128; j += 256) {
    const int m = j >> 7, c = j & 127;
    float f = sNbr[m][0] * w1[c * 3 + 0] + sNbr[m][1] * w1[c * 3 + 1] +
              sNbr[m][2] * w1[c * 3 + 2] + b1[c];
    const float sc = g1[c] * rsqrtf(v1[c] + BN_EPS);
    f = (f - m1[c]) * sc + be1[c];
    sA1[m * STR1 + c] = (_Float16)fmaxf(f, 0.0f);
  }
  __syncthreads();

  // layer2 (WMMA): [64,128] x [128,256] + b2; per-group colmax -> feat=[fg|f2]
  {
    const int ntBase = 2 * wave;                // 2 n-tiles per wave
    Acc8 acc[2][4];
    gemm_block<2, 4>(sA1, STR1, w2h, 128, ntBase, lane, acc);
#pragma unroll
    for (int j = 0; j < 2; ++j) {
      const int ch = (ntBase + j) * 16 + (lane & 15);
      const float b2v = b2[ch];
      float cm0 = -3.4e38f, cm1 = -3.4e38f;
#pragma unroll
      for (int mt = 0; mt < 4; ++mt) {
#pragma unroll
        for (int v = 0; v < 8; ++v) {
          const float f2 = acc[j][mt].f[v] + b2v;
          const int m = mt * 16 + lg * 8 + v;
          sFeat[m * STRF + 256 + ch] = (_Float16)f2;
          if (mt < 2) cm0 = fmaxf(cm0, f2); else cm1 = fmaxf(cm1, f2);
        }
      }
      cm0 = fmaxf(cm0, __shfl_xor(cm0, 16, 32));
      cm1 = fmaxf(cm1, __shfl_xor(cm1, 16, 32));
      const _Float16 h0 = (_Float16)cm0, h1 = (_Float16)cm1;
      const int mBase = lg * 16;
      for (int m = 0; m < 16; ++m) {
        sFeat[(mBase + m) * STRF + ch]      = h0;   // group A fg
        sFeat[(32 + mBase + m) * STRF + ch] = h1;   // group B fg
      }
    }
  }
  __syncthreads();

  // layer3 (WMMA): [64,512] x [512,512] + b3, BN + ReLU
  {
    const int ntBase = 4 * wave;                // 4 n-tiles per wave
    Acc8 acc[4][4];
    gemm_block<4, 16>(sFeat, STRF, w3h, 512, ntBase, lane, acc);
#pragma unroll
    for (int j = 0; j < 4; ++j) {
      const int ch = (ntBase + j) * 16 + (lane & 15);
      const float sc = g3[ch] * rsqrtf(v3[ch] + BN_EPS);
      const float sh = be3[ch] - m3[ch] * sc;
      const float bb = b3[ch];
#pragma unroll
      for (int mt = 0; mt < 4; ++mt) {
#pragma unroll
        for (int v = 0; v < 8; ++v) {
          const float z = fmaxf((acc[j][mt].f[v] + bb) * sc + sh, 0.0f);
          const int m = mt * 16 + lg * 8 + v;
          sA3[m * STRF + ch] = (_Float16)z;
        }
      }
    }
  }
  __syncthreads();

  // layer4 (WMMA): [64,512] x [512,384]; tokens = per-group colmax + b4
  {
    const int ntBase = 3 * wave;                // 3 n-tiles per wave
    Acc8 acc[3][4];
    gemm_block<3, 16>(sA3, STRF, w4h, 512, ntBase, lane, acc);
#pragma unroll
    for (int j = 0; j < 3; ++j) {
      const int ch = (ntBase + j) * 16 + (lane & 15);
      float cm0 = -3.4e38f, cm1 = -3.4e38f;
#pragma unroll
      for (int mt = 0; mt < 4; ++mt)
#pragma unroll
        for (int v = 0; v < 8; ++v) {
          if (mt < 2) cm0 = fmaxf(cm0, acc[j][mt].f[v]);
          else        cm1 = fmaxf(cm1, acc[j][mt].f[v]);
        }
      cm0 = fmaxf(cm0, __shfl_xor(cm0, 16, 32)) + b4[ch];
      cm1 = fmaxf(cm1, __shfl_xor(cm1, 16, 32)) + b4[ch];
      if (lg == 0 && lane < 16) {
        tokens[(size_t)gA * 384 + ch]       = cm0;
        tokens[(size_t)(gA + 1) * 384 + ch] = cm1;
      }
    }
  }
}

// ---- Host-side launcher -----------------------------------------------------
extern "C" void kernel_launch(void* const* d_in, const int* in_sizes, int n_in,
                              void* d_out, int out_size, void* d_ws,
                              size_t ws_size, hipStream_t stream) {
  const float* xyz = (const float*)d_in[0];
  const float* w1  = (const float*)d_in[1];
  const float* b1  = (const float*)d_in[2];
  const float* g1  = (const float*)d_in[3];
  const float* be1 = (const float*)d_in[4];
  const float* m1  = (const float*)d_in[5];
  const float* v1  = (const float*)d_in[6];
  const float* w2  = (const float*)d_in[7];
  const float* b2  = (const float*)d_in[8];
  const float* w3  = (const float*)d_in[9];
  const float* b3  = (const float*)d_in[10];
  const float* g3  = (const float*)d_in[11];
  const float* be3 = (const float*)d_in[12];
  const float* m3  = (const float*)d_in[13];
  const float* v3  = (const float*)d_in[14];
  const float* w4  = (const float*)d_in[15];
  const float* b4  = (const float*)d_in[16];

  float* out    = (float*)d_out;
  float* center = out;                       // [16,512,3]
  float* tokens = out + 16 * NG * 3;         // [16,512,384]

  char* ws = (char*)d_ws;
  int*      fidx = (int*)ws;                               // 32 KB
  int*      knn  = (int*)(ws + 32768);                     // 1 MB
  _Float16* w2h  = (_Float16*)(ws + 32768 + 1048576);      // 64 KB
  _Float16* w3h  = w2h + 256 * 128;                        // 512 KB
  _Float16* w4h  = w3h + 512 * 512;                        // 384 KB

  convert_weights<<<1024, 256, 0, stream>>>(w2, w3, w4, w2h, w3h, w4h);
  fps_kernel<<<16, 256, 0, stream>>>(xyz, fidx, center);
  knn_kernel<<<16 * NG, 256, 0, stream>>>(xyz, center, knn);
  sa_mlp_kernel<<<16 * NG / 2, 256, 0, stream>>>(
      xyz, center, knn, w1, b1, g1, be1, m1, v1, b2, b3, g3, be3, m3, v3, b4,
      w2h, w3h, w4h, tokens);
}
